// RandomLayer_82944408420426
// MI455X (gfx1250) — compile-verified, matrix operand order checked by reading, not run on hardware
//
#include <hip/hip_runtime.h>

// Problem constants (from the reference)
#define D_MODEL    2048
#define N_CLASSES  100
#define OUT_DIM    1024
#define BATCH      512

// Tiling
#define BLOCK_M    128
#define BLOCK_N    128
#define SPLIT_K    16
#define K_TOTAL    (D_MODEL * N_CLASSES)       // 204800
#define K_CHUNK    (K_TOTAL / SPLIT_K)         // 12800 = 128 i-columns exactly
#define K_ROUND    64                          // staged K per round (2 WMMA slices)
#define NROUNDS    (K_CHUNK / K_ROUND)         // 200

// LDS layout
#define AB_STRIDE  72    // bf16 per row: 64 data + 8 pad -> 144B rows, 16B aligned
#define S_STRIDE   102   // f32, even -> 8B-aligned pair (b64) reads

typedef __attribute__((ext_vector_type(16))) __bf16 v16bf;
typedef __attribute__((ext_vector_type(8)))  __bf16 v8bf;
typedef __attribute__((ext_vector_type(2)))  __bf16 v2bf;
typedef __attribute__((ext_vector_type(2)))  float  v2f;
typedef __attribute__((ext_vector_type(8)))  float  v8f;

__global__ void zero_out_kernel(float* __restrict__ p, int n) {
    int i = blockIdx.x * blockDim.x + threadIdx.x;
    if (i < n) p[i] = 0.0f;
}

// 16x32 bf16 fragment from LDS (A layout; B mirrored, tile stored N-major).
// CDNA5 ISA 7.12.2: lane half 0: K = 0..7 & 16..23 ; half 1: K = 8..15 & 24..31
__device__ inline v16bf load_frag(const __bf16* __restrict__ row_base, int kh) {
    v8bf lo = *(const v8bf*)(row_base + kh * 8);
    v8bf hi = *(const v8bf*)(row_base + 16 + kh * 8);
    return __builtin_shufflevector(lo, hi,
        0, 1, 2, 3, 4, 5, 6, 7, 8, 9, 10, 11, 12, 13, 14, 15);
}

__global__ __launch_bounds__(256)
void fused_kr_gemm(const float* __restrict__ features,   // [512, 2048]
                   const float* __restrict__ softmax,    // [512, 100]
                   const float* __restrict__ R,          // [204800, 1024]
                   float* __restrict__ out)              // [512, 1024]
{
    __shared__ float s_cache[BLOCK_M * S_STRIDE];                    // 52.2 KB
    __shared__ __align__(16) __bf16 A_lds[BLOCK_M * AB_STRIDE];      // 18.4 KB
    __shared__ __align__(16) __bf16 B_lds[BLOCK_N * AB_STRIDE];      // 18.4 KB

    const int tid  = threadIdx.x;
    const int lane = tid & 31;
    const int wid  = tid >> 5;
    const int kh   = lane >> 4;   // lane half (K-split of fragment)
    const int l16  = lane & 15;   // row (A: M, B: N) within 16x16 subtile

    const int m0 = blockIdx.x * BLOCK_M;   // M fastest -> B-slab L2 reuse
    const int n0 = blockIdx.y * BLOCK_N;
    const int kc = blockIdx.z * K_CHUNK;

    // Wave tile: 32(M) x 64(N)  => 2x4 accumulators of 16x16
    const int wm = wid >> 1;   // 0..3
    const int wn = wid & 1;    // 0..1

    // ---- Cache this block's softmax rows once
    for (int idx = tid; idx < BLOCK_M * N_CLASSES; idx += 256) {
        int b = idx / N_CLASSES;
        int j = idx - b * N_CLASSES;
        s_cache[b * S_STRIDE + j] = softmax[(m0 + b) * N_CLASSES + j];
    }
    __syncthreads();

    v8f acc[2][4] = {};

    const int    bA   = tid & 127;                    // fixed batch row for A staging
    const size_t frow = (size_t)(m0 + bA) * D_MODEL;

    // Software pipeline registers: next round's R data + feature scalars
    float4 breg[8];
    float  f_lo, f_hi;

    // ---- Prefetch round 0
    {
        const int kbase = kc;
        const int i_lo = kbase / N_CLASSES;
        const int i_hi = (kbase + K_ROUND - 1) / N_CLASSES;   // span 64 < 100 -> <= i_lo+1
        f_lo = features[frow + i_lo];
        f_hi = features[frow + i_hi];
        #pragma unroll
        for (int it = 0; it < 4; ++it) {
            int u  = tid + it * 256;          // 1024 units: (k-pair, n4)
            int kp = u >> 5, n4 = u & 31;
            const float* src = R + (size_t)(kbase + 2 * kp) * OUT_DIM + n0 + n4 * 4;
            breg[2 * it]     = *(const float4*)(src);
            breg[2 * it + 1] = *(const float4*)(src + OUT_DIM);
        }
    }

    for (int ks = 0; ks < NROUNDS; ++ks) {
        const int kbase = kc + ks * K_ROUND;
        const int i_lo  = kbase / N_CLASSES;
        const float fl = f_lo, fh = f_hi;

        // ---- Stage B from prefetch regs: packed bf16 pairs, K-transposed
        #pragma unroll
        for (int it = 0; it < 4; ++it) {
            int u  = tid + it * 256;
            int kp = u >> 5, n4 = u & 31;
            const float4 a = breg[2 * it];
            const float4 b = breg[2 * it + 1];
            int n = n4 * 4, kk = 2 * kp;      // kk even -> 4B-aligned store
            *(v2bf*)&B_lds[(n + 0) * AB_STRIDE + kk] = (v2bf){(__bf16)a.x, (__bf16)b.x};
            *(v2bf*)&B_lds[(n + 1) * AB_STRIDE + kk] = (v2bf){(__bf16)a.y, (__bf16)b.y};
            *(v2bf*)&B_lds[(n + 2) * AB_STRIDE + kk] = (v2bf){(__bf16)a.z, (__bf16)b.z};
            *(v2bf*)&B_lds[(n + 3) * AB_STRIDE + kk] = (v2bf){(__bf16)a.w, (__bf16)b.w};
        }

        // ---- Generate A tile in even-kk pairs:
        // k even => j = k%100 even => (j, j+1) contiguous, same i. One b64 softmax
        // read + one packed b32 store per pair.
        #pragma unroll
        for (int it = 0; it < 16; ++it) {
            int u  = tid + it * 256;
            int pp = u >> 7;                  // pair index 0..31 (b = bA fixed)
            int kk = 2 * pp;
            int k  = kbase + kk;
            int i  = k / N_CLASSES;
            int j  = k - i * N_CLASSES;       // even
            float fv = (i == i_lo) ? fl : fh;
            v2f sp = *(const v2f*)&s_cache[bA * S_STRIDE + j];
            *(v2bf*)&A_lds[bA * AB_STRIDE + kk] =
                (v2bf){(__bf16)(fv * sp.x), (__bf16)(fv * sp.y)};
        }
        __syncthreads();

        // ---- Prefetch next round (global loads overlap the WMMA phase)
        if (ks + 1 < NROUNDS) {
            const int kn   = kbase + K_ROUND;
            const int ni_l = kn / N_CLASSES;
            const int ni_h = (kn + K_ROUND - 1) / N_CLASSES;
            f_lo = features[frow + ni_l];
            f_hi = features[frow + ni_h];
            #pragma unroll
            for (int it = 0; it < 4; ++it) {
                int u  = tid + it * 256;
                int kp = u >> 5, n4 = u & 31;
                const float* src = R + (size_t)(kn + 2 * kp) * OUT_DIM + n0 + n4 * 4;
                breg[2 * it]     = *(const float4*)(src);
                breg[2 * it + 1] = *(const float4*)(src + OUT_DIM);
            }
        }

        // ---- Compute: two 32-wide K slices, 8 WMMAs each
        #pragma unroll
        for (int s = 0; s < 2; ++s) {
            v16bf afrag[2], bfrag[4];
            #pragma unroll
            for (int mi = 0; mi < 2; ++mi)
                afrag[mi] = load_frag(
                    A_lds + (wm * 32 + mi * 16 + l16) * AB_STRIDE + s * 32, kh);
            #pragma unroll
            for (int ni = 0; ni < 4; ++ni)
                bfrag[ni] = load_frag(
                    B_lds + (wn * 64 + ni * 16 + l16) * AB_STRIDE + s * 32, kh);
            #pragma unroll
            for (int mi = 0; mi < 2; ++mi)
                #pragma unroll
                for (int ni = 0; ni < 4; ++ni)
                    acc[mi][ni] = __builtin_amdgcn_wmma_f32_16x16x32_bf16(
                        false, afrag[mi], false, bfrag[ni],
                        (short)0, acc[mi][ni], false, false);
        }
        __syncthreads();   // protect LDS tiles against next round's staging
    }

    // ---- Epilogue: split-K partials accumulate into d_out.
    // C/D layout: VGPR r, lanes 0-15 -> M = r, lanes 16-31 -> M = 8 + r; N = l16.
    #pragma unroll
    for (int mi = 0; mi < 2; ++mi) {
        const int mrow = m0 + wm * 32 + mi * 16 + kh * 8;
        #pragma unroll
        for (int ni = 0; ni < 4; ++ni) {
            const int n = n0 + wn * 64 + ni * 16 + l16;
            #pragma unroll
            for (int r = 0; r < 8; ++r)
                atomicAdd(&out[(size_t)(mrow + r) * OUT_DIM + n], acc[mi][ni][r]);
        }
    }
}

extern "C" void kernel_launch(void* const* d_in, const int* in_sizes, int n_in,
                              void* d_out, int out_size, void* d_ws, size_t ws_size,
                              hipStream_t stream) {
    const float* features = (const float*)d_in[0];   // [512, 2048]
    const float* softmax  = (const float*)d_in[1];   // [512, 100]
    const float* R        = (const float*)d_in[2];   // [204800, 1024]
    float* out = (float*)d_out;                      // [512, 1024]

    const int n_out = BATCH * OUT_DIM;
    zero_out_kernel<<<(n_out + 255) / 256, 256, 0, stream>>>(out, n_out);

    dim3 grid(BATCH / BLOCK_M, OUT_DIM / BLOCK_N, SPLIT_K);  // (4, 8, 16)
    fused_kr_gemm<<<grid, 256, 0, stream>>>(features, softmax, R, out);
}
